// ProcessModule_18502719111642
// MI455X (gfx1250) — compile-verified, matrix operand order checked by reading, not run on hardware
//
#include <hip/hip_runtime.h>
#include <hip/hip_bf16.h>

typedef __attribute__((ext_vector_type(16))) _Float16 v16h;
typedef __attribute__((ext_vector_type(8)))  _Float16 v8h;
typedef __attribute__((ext_vector_type(4)))  _Float16 v4h;
typedef __attribute__((ext_vector_type(8)))  float    v8f;

#define MAXD 12
#define H 128

// K index inside a 16x32 f16 WMMA fragment for element j (0..15), per CDNA5 ISA 7.12.2:
// lanes 0-15 (half=0): VGPR0-3 -> K 0..7, VGPR4-7 -> K 16..23 ; lanes 16-31: K 8..15 / 24..31.
// For j in 0..7:  k = half*8 + j ; for j in 8..15: k = 16 + half*8 + (j-8).
__device__ __forceinline__ int kmap(int j, int half) {
  return (((j >> 3) << 1) + half) * 8 + (((j >> 1) & 3) << 1) + (j & 1);
}

// A (or hidden) fragment from an LDS row: two contiguous 8-half runs -> 2x ds_load_b128.
__device__ __forceinline__ v16h load_frag_lds(const _Float16* rowbase, int kk, int half) {
  const v8h a0 = *(const v8h*)(rowbase + kk + half * 8);
  const v8h a1 = *(const v8h*)(rowbase + kk + 16 + half * 8);
  v16h r;
#pragma unroll
  for (int j = 0; j < 8; ++j) { r[j] = a0[j]; r[j + 8] = a1[j]; }
  return r;
}

// B fragment from fragment-ordered weights: 32 contiguous halves per (ktile,col),
// split by half-wave -> 2x global_load_b128.
__device__ __forceinline__ v16h load_frag_w(const _Float16* Wf, int ktile, int col, int half) {
  const _Float16* p = Wf + ((((ktile << 7) + col) << 5) + (half << 4));
  const v8h b0 = *(const v8h*)(p);
  const v8h b1 = *(const v8h*)(p + 8);
  v16h r;
#pragma unroll
  for (int j = 0; j < 8; ++j) { r[j] = b0[j]; r[j + 8] = b1[j]; }
  return r;
}

// ---------------- small utility kernels ----------------

__global__ void copy_f32(const float* __restrict__ src, float* __restrict__ dst, int n) {
  int i = blockIdx.x * blockDim.x + threadIdx.x;
  if (i < n) dst[i] = src[i];
}

// Convert W [K][128] f32 -> f16 in WMMA B-fragment order:
// Wf[ ktile ][ n ][ half ][ j ] = W[ ktile*32 + kmap(j,half) ][ n ]  (zero-padded K).
__global__ void prep_weights_frag(const float* __restrict__ W, _Float16* __restrict__ Wf,
                                  int K, int KPAD) {
  int idx = blockIdx.x * blockDim.x + threadIdx.x;
  if (idx >= KPAD * H) return;
  int j = idx & 15;
  int half = (idx >> 4) & 1;
  int n = (idx >> 5) & 127;
  int ktile = idx >> 12;
  int k = ktile * 32 + kmap(j, half);
  float v = (k < K) ? W[k * H + n] : 0.0f;
  Wf[idx] = (_Float16)v;
}

__global__ void zero_flags(int* __restrict__ flags, int* __restrict__ counters, int n) {
  int i = blockIdx.x * blockDim.x + threadIdx.x;
  if (i < 8) counters[i] = 0;
  if (i < n) flags[i] = 0;
}

// One thread per node: classify children at this depth, build child lists,
// set parent flag bits, and zero the scatter-target rows (LR + MH) of the parent.
__global__ void classify(const int* __restrict__ depths, const int* __restrict__ states,
                         const int* __restrict__ parents, int depth, int n,
                         int* __restrict__ flags, int* __restrict__ counters,
                         int* __restrict__ listL, int* __restrict__ listR,
                         int* __restrict__ listH,
                         float* __restrict__ LR, float* __restrict__ MH) {
  int i = blockIdx.x * blockDim.x + threadIdx.x;
  if (i >= n) return;
  if (depths[i] != depth) return;
  int s = states[i];
  if (s != 0 && s != 1 && s != 3) return;
  int p = parents[i];
  if (s == 0)      { listL[atomicAdd(&counters[0], 1)] = i; atomicOr(&flags[p], 1); }
  else if (s == 1) { listR[atomicAdd(&counters[1], 1)] = i; }
  else             { listH[atomicAdd(&counters[2], 1)] = i; atomicOr(&flags[p], 2); }
  // zero left/right (LR row = 256 f32) and merged_heads (128 f32) of parent.
  // Races between siblings are benign (everyone writes zeros).
  float4 z = make_float4(0.f, 0.f, 0.f, 0.f);
  float4* lr = (float4*)(LR + (size_t)p * 256);
#pragma unroll 4
  for (int q = 0; q < 64; ++q) lr[q] = z;
  float4* mh = (float4*)(MH + (size_t)p * H);
#pragma unroll 4
  for (int q = 0; q < 32; ++q) mh[q] = z;
}

__global__ void build_parent_lists(const int* __restrict__ flags, int n,
                                   int* __restrict__ counters,
                                   int* __restrict__ listP, int* __restrict__ listD) {
  int i = blockIdx.x * blockDim.x + threadIdx.x;
  if (i >= n) return;
  int f = flags[i];
  if (!f) return;
  if (f & 1) listP[atomicAdd(&counters[3], 1)] = i;
  if (f & 2) listD[atomicAdd(&counters[4], 1)] = i;
}

// 32 lanes per active child row: atomic scatter-add x[child] into LR[parent].
__global__ void scatter_lr(const int* __restrict__ listL, const int* __restrict__ listR,
                           const int* __restrict__ counters, const int* __restrict__ parents,
                           const float* __restrict__ xcur, float* __restrict__ LR) {
  int gid = blockIdx.x * blockDim.x + threadIdx.x;
  int idx = gid >> 5;
  int lane = gid & 31;
  int cL = counters[0], cR = counters[1];
  if (idx >= cL + cR) return;
  int child, off;
  if (idx < cL) { child = listL[idx];      off = 0; }
  else          { child = listR[idx - cL]; off = H; }
  int p = parents[child];
  float* dst = LR + (size_t)p * 256 + off;
  const float* src = xcur + (size_t)child * H;
  int c = lane * 4;
  atomicAdd(dst + c + 0, src[c + 0]);
  atomicAdd(dst + c + 1, src[c + 1]);
  atomicAdd(dst + c + 2, src[c + 2]);
  atomicAdd(dst + c + 3, src[c + 3]);
}

// ---------------- fused gather -> (Linear,ReLU,Linear) -> scatter via WMMA ----------------
// MODE 0: lep     in = [x(i) | plef(i)]      KIN=144 KPAD=160  out: atomicAdd MH[parents[i]]
// MODE 1: lem     in = [x(p) | MH(p)]        KIN=256 KPAD=256  out: x[p] = result
// MODE 2: merger  in = [LR(p) | pef(p)]      KIN=272 KPAD=288  out: x[p] = result
// One workgroup = 16 rows, 8 waves, each wave owns a 16-column tile of the 128 outputs.
template <int KIN, int KPAD, int MODE>
__global__ __launch_bounds__(256)
void mlp2_wmma(const int* __restrict__ rowlist, const int* __restrict__ cnt_ptr,
               const float* __restrict__ in0, const float* __restrict__ in1,
               const _Float16* __restrict__ W1f, const float* __restrict__ b1,
               const _Float16* __restrict__ W2f, const float* __restrict__ b2,
               float* __restrict__ out_x, float* __restrict__ out_mh,
               const int* __restrict__ parents) {
  const int cnt = *cnt_ptr;
  const int tile = blockIdx.x;
  if (tile * 16 >= cnt) return;

  __shared__ alignas(16) _Float16 sA[16 * KPAD];
  __shared__ alignas(16) _Float16 sH[16 * H];
  __shared__ int sRow[16];
  __shared__ int sPar[16];

  const int t = threadIdx.x;
  if (t < 16) {
    int g = tile * 16 + t;
    int r = (g < cnt) ? rowlist[g] : -1;
    sRow[t] = r;
    sPar[t] = (MODE == 0 && r >= 0) ? parents[r] : -1;
  }
  __syncthreads();

  // gather + f32->f16 convert into LDS, 4 halves per thread iteration.
  // Region boundaries (128/144/256/272/KIN) are multiples of 4, so chunks never straddle.
  for (int q = t; q < (16 * KPAD) / 4; q += 256) {
    int idx = q * 4;
    int r = idx / KPAD;
    int k = idx - r * KPAD;
    int row = sRow[r];
    float4 v = make_float4(0.f, 0.f, 0.f, 0.f);
    if (row >= 0) {
      if (MODE == 0) {
        if (k < H)        v = *(const float4*)(in0 + (size_t)row * H + k);
        else if (k < KIN) v = *(const float4*)(in1 + (size_t)row * 16 + (k - H));
      } else if (MODE == 1) {
        if (k < H)        v = *(const float4*)(in0 + (size_t)row * H + k);
        else              v = *(const float4*)(in1 + (size_t)row * H + (k - H));
      } else {
        if (k < 256)      v = *(const float4*)(in0 + (size_t)row * 256 + k);
        else if (k < KIN) v = *(const float4*)(in1 + (size_t)row * 16 + (k - 256));
      }
    }
    v4h h;
    h[0] = (_Float16)v.x; h[1] = (_Float16)v.y; h[2] = (_Float16)v.z; h[3] = (_Float16)v.w;
    *(v4h*)(sA + idx) = h;
  }
  __syncthreads();

  const int lane = t & 31;
  const int half = lane >> 4;
  const int nl = lane & 15;            // A-row (M) / D-column (N) within tile
  const int colbase = (t >> 5) * 16;   // this wave's 16-column slice
  const int col = colbase + nl;

  // warm L2/L0 with this wave's layer-2 weight fragments while layer 1 runs
  __builtin_prefetch(W2f + ((size_t)col << 5), 0, 0);
  __builtin_prefetch(W2f + (((size_t)H + col) << 5), 0, 0);

  // ---- layer 1: hidden = relu(A @ W1 + b1) ----
  v8f c;
  {
    float bias = b1[col];
#pragma unroll
    for (int r = 0; r < 8; ++r) c[r] = bias;
  }
#pragma unroll
  for (int kt = 0; kt < KPAD / 32; ++kt) {
    v16h a = load_frag_lds(sA + nl * KPAD, kt * 32, half);
    v16h b = load_frag_w(W1f, kt, col, half);
    c = __builtin_amdgcn_wmma_f32_16x16x32_f16(false, a, false, b, (short)0, c, false, false);
  }
#pragma unroll
  for (int r = 0; r < 8; ++r) {
    float v = c[r] > 0.0f ? c[r] : 0.0f;
    int m = r + 8 * half;
    sH[m * H + col] = (_Float16)v;
  }
  __syncthreads();

  // ---- layer 2: out = hidden @ W2 + b2 ----
  {
    float bias = b2[col];
#pragma unroll
    for (int r = 0; r < 8; ++r) c[r] = bias;
  }
#pragma unroll
  for (int kt = 0; kt < H / 32; ++kt) {
    v16h a = load_frag_lds(sH + nl * H, kt * 32, half);
    v16h b = load_frag_w(W2f, kt, col, half);
    c = __builtin_amdgcn_wmma_f32_16x16x32_f16(false, a, false, b, (short)0, c, false, false);
  }

#pragma unroll
  for (int r = 0; r < 8; ++r) {
    int m = r + 8 * half;
    int row = sRow[m];
    if (row < 0) continue;
    if (MODE == 0) {
      atomicAdd(out_mh + (size_t)sPar[m] * H + col, c[r]);
    } else {
      out_x[(size_t)row * H + col] = c[r];
    }
  }
}

// ---------------- host launcher ----------------

extern "C" void kernel_launch(void* const* d_in, const int* in_sizes, int n_in,
                              void* d_out, int out_size, void* d_ws, size_t ws_size,
                              hipStream_t stream) {
  const int N = in_sizes[2];  // depths has N elements

  const float* x       = (const float*)d_in[0];
  const int* edgeidx   = (const int*)d_in[1];
  const int* parents   = edgeidx + N;        // edge_index[1]
  const int* depths    = (const int*)d_in[2];
  const int* states    = (const int*)d_in[3];
  const float* pef     = (const float*)d_in[4];
  const float* plef    = (const float*)d_in[5];
  const float* mW1 = (const float*)d_in[6];  const float* mb1 = (const float*)d_in[7];
  const float* mW2 = (const float*)d_in[8];  const float* mb2 = (const float*)d_in[9];
  const float* pW1 = (const float*)d_in[10]; const float* pb1 = (const float*)d_in[11];
  const float* pW2 = (const float*)d_in[12]; const float* pb2 = (const float*)d_in[13];
  const float* eW1 = (const float*)d_in[14]; const float* eb1 = (const float*)d_in[15];
  const float* eW2 = (const float*)d_in[16]; const float* eb2 = (const float*)d_in[17];

  float* xcur = (float*)d_out;               // x state lives in d_out, updated in place

  // workspace carve-up
  float* LR     = (float*)d_ws;                        // [N][256] left|right
  float* MH     = LR + (size_t)N * 256;                // [N][128] merged_heads
  int* flags    = (int*)(MH + (size_t)N * H);          // [N]
  int* counters = flags + N;                           // [8]: L,R,H,P,D
  int* listL    = counters + 8;
  int* listR    = listL + N;
  int* listH    = listR + N;
  int* listP    = listH + N;
  int* listD    = listP + N;
  _Float16* W1mf = (_Float16*)(listD + N);             // merger W1: 288x128 (frag order)
  _Float16* W2mf = W1mf + 288 * H;                     // merger W2: 128x128
  _Float16* W1lf = W2mf + H * H;                       // lep W1: 160x128
  _Float16* W2lf = W1lf + 160 * H;                     // lep W2
  _Float16* W1ef = W2lf + H * H;                       // lem W1: 256x128
  _Float16* W2ef = W1ef + 256 * H;                     // lem W2

  const int TPB = 256;
  const int gN = (N + TPB - 1) / TPB;

  // one-time per launch: f32->f16 fragment-ordered weights, x -> xcur
  prep_weights_frag<<<(288 * H + TPB - 1) / TPB, TPB, 0, stream>>>(mW1, W1mf, 272, 288);
  prep_weights_frag<<<(H * H + TPB - 1) / TPB, TPB, 0, stream>>>(mW2, W2mf, H, H);
  prep_weights_frag<<<(160 * H + TPB - 1) / TPB, TPB, 0, stream>>>(pW1, W1lf, 144, 160);
  prep_weights_frag<<<(H * H + TPB - 1) / TPB, TPB, 0, stream>>>(pW2, W2lf, H, H);
  prep_weights_frag<<<(256 * H + TPB - 1) / TPB, TPB, 0, stream>>>(eW1, W1ef, 256, 256);
  prep_weights_frag<<<(H * H + TPB - 1) / TPB, TPB, 0, stream>>>(eW2, W2ef, H, H);
  copy_f32<<<((N * H) + TPB - 1) / TPB, TPB, 0, stream>>>(x, xcur, N * H);

  const int gTiles = (N + 15) / 16;          // worst-case; blocks early-exit on count
  const int gScat  = (N * 32 + TPB - 1) / TPB;

  for (int depth = MAXD; depth >= 1; --depth) {
    zero_flags<<<gN, TPB, 0, stream>>>(flags, counters, N);
    classify<<<gN, TPB, 0, stream>>>(depths, states, parents, depth, N,
                                     flags, counters, listL, listR, listH, LR, MH);
    build_parent_lists<<<gN, TPB, 0, stream>>>(flags, N, counters, listP, listD);
    // lep on head children -> atomic scatter into merged_heads (reads xcur before any update)
    mlp2_wmma<144, 160, 0><<<gTiles, TPB, 0, stream>>>(
        listH, counters + 2, xcur, plef, W1lf, pb1, W2lf, pb2, nullptr, MH, parents);
    // left/right scatter sums (reads xcur before any update)
    scatter_lr<<<gScat, TPB, 0, stream>>>(listL, listR, counters, parents, xcur, LR);
    // lem on designated parents -> writes xcur
    mlp2_wmma<256, 256, 1><<<gTiles, TPB, 0, stream>>>(
        listD, counters + 4, xcur, MH, W1ef, eb1, W2ef, eb2, xcur, nullptr, parents);
    // merger on parents_mask rows -> writes xcur last (overrides lem, per reference select)
    mlp2_wmma<272, 288, 2><<<gTiles, TPB, 0, stream>>>(
        listP, counters + 3, LR, pef, W1mf, mb1, W2mf, mb2, xcur, nullptr, parents);
  }
}
